// SS1D_83769042141513
// MI455X (gfx1250) — matrix operand order repaired and checked by reading, not compiled, not run on hardware
//
#include <hip/hip_runtime.h>
#include <hip/hip_bf16.h>
#include <math.h>

// Problem constants
#define B_   8
#define L_   2048
#define DM   192
#define DI   384
#define N_   16
#define R_   12
#define M_   (B_ * L_)      // 16384 rows
#define NC1  (2 * DI)       // 768 cols of GEMM1
#define DBL_LD 48           // padded leading dim for dbl (44 -> 48)
#define NCH  32             // scan chunks per batch
#define CHL  (L_ / NCH)     // 64 timesteps per chunk
#define LN_EPS 1e-5f

typedef __bf16 bf16;
typedef __attribute__((ext_vector_type(16))) __bf16 v16bf;
typedef __attribute__((ext_vector_type(8)))  __bf16 bf16x8;
typedef __attribute__((ext_vector_type(8)))  float  v8f;

// Fast sigmoid: one v_exp_f32 + one v_rcp_f32 (both TRANS, co-execute with VALU)
__device__ __forceinline__ float fast_sigmoid(float v) {
  return __builtin_amdgcn_rcpf(1.0f + __expf(-v));
}

// ---------------------------------------------------------------------------
// Prep: fp32 -> bf16 weight/activation conversion, A = -exp(A_logs),
// zero-padded x_proj (44 -> 48 rows).
// ---------------------------------------------------------------------------
__global__ void ss1d_prep(const float* __restrict__ x, const float* __restrict__ W_in,
                          const float* __restrict__ x_proj_w, const float* __restrict__ A_logs,
                          const float* __restrict__ W_out,
                          bf16* __restrict__ x_bf, bf16* __restrict__ win_bf,
                          bf16* __restrict__ xproj_bf, float* __restrict__ Aneg,
                          bf16* __restrict__ wout_bf) {
  int stride = gridDim.x * blockDim.x;
  for (int idx = blockIdx.x * blockDim.x + threadIdx.x; idx < M_ * DM; idx += stride) {
    x_bf[idx] = (bf16)x[idx];
    if (idx < NC1 * DM) win_bf[idx] = (bf16)W_in[idx];
    if (idx < 48 * DI) {
      int r = idx / DI;
      xproj_bf[idx] = (r < (R_ + 2 * N_)) ? (bf16)x_proj_w[idx] : (bf16)0.0f;
    }
    if (idx < DI * N_) Aneg[idx] = -__expf(A_logs[idx]);
    if (idx < DM * DI) wout_bf[idx] = (bf16)W_out[idx];
  }
}

// ---------------------------------------------------------------------------
// One 16x16 output tile via v_wmma_f32_16x16x32_bf16, K-loop over 32.
// A: row-major (row = m0 + lane&15).  ISA layout: lane<16 K={k0..k0+7, k0+16..23},
//    lane>=16 K shifted by +8.
// B: element [K][N] at Bm[(n0+lane&15)*ldb + K]; same K layout as A.
// ---------------------------------------------------------------------------
__device__ __forceinline__ v8f wmma_bf16_tile(const bf16* __restrict__ A, int lda,
                                              const bf16* __restrict__ Bm, int ldb,
                                              int m0, int n0, int K) {
  const int lane = threadIdx.x & 31;
  const int hi = (lane >= 16) ? 1 : 0;
  const bf16* __restrict__ Arow = A + (size_t)(m0 + (lane & 15)) * lda;
  const bf16* __restrict__ Bcol = Bm + (size_t)(n0 + (lane & 15)) * ldb;
  v8f acc = {0.f, 0.f, 0.f, 0.f, 0.f, 0.f, 0.f, 0.f};
  for (int k0 = 0; k0 < K; k0 += 32) {
    bf16x8 a0 = *(const bf16x8*)(Arow + k0 + hi * 8);
    bf16x8 a1 = *(const bf16x8*)(Arow + k0 + 16 + hi * 8);
    bf16x8 b0 = *(const bf16x8*)(Bcol + k0 + hi * 8);
    bf16x8 b1 = *(const bf16x8*)(Bcol + k0 + 16 + hi * 8);
    v16bf av, bv;
#pragma unroll
    for (int j = 0; j < 8; ++j) {
      av[j] = a0[j]; av[j + 8] = a1[j];
      bv[j] = b0[j]; bv[j + 8] = b1[j];
    }
    acc = __builtin_amdgcn_wmma_f32_16x16x32_bf16(false, av, false, bv,
                                                  (short)0, acc, false, false);
  }
  return acc;
}

// GEMM1: xz = silu(x @ W_in^T); split -> u (f32 + bf16 copy), z (f32)
__global__ void ss1d_gemm1(const bf16* __restrict__ x_bf, const bf16* __restrict__ win_bf,
                           float* __restrict__ u_f, bf16* __restrict__ u_bf,
                           float* __restrict__ z_f) {
  const int NT = NC1 / 16;  // 48 n-tiles
  int wid = (blockIdx.x * blockDim.x + threadIdx.x) >> 5;
  if (wid >= (M_ / 16) * NT) return;
  int mt = wid / NT, nt = wid % NT;
  v8f acc = wmma_bf16_tile(x_bf, DM, win_bf, DM, mt * 16, nt * 16, DM);
  const int lane = threadIdx.x & 31;
  const int col = nt * 16 + (lane & 15);
  const int rbase = (lane < 16) ? 0 : 8;
#pragma unroll
  for (int r = 0; r < 8; ++r) {
    int m = mt * 16 + rbase + r;
    float v = acc[r];
    float s = v * fast_sigmoid(v);  // silu via v_exp + v_rcp
    if (col < DI) {
      u_f[(size_t)m * DI + col] = s;
      u_bf[(size_t)m * DI + col] = (bf16)s;
    } else {
      z_f[(size_t)m * DI + (col - DI)] = s;
    }
  }
}

// GEMM2: dbl = u @ x_proj_w^T  (N=44 padded to 48)
__global__ void ss1d_gemm2(const bf16* __restrict__ u_bf, const bf16* __restrict__ xproj_bf,
                           float* __restrict__ dbl) {
  const int NT = DBL_LD / 16;  // 3
  int wid = (blockIdx.x * blockDim.x + threadIdx.x) >> 5;
  if (wid >= (M_ / 16) * NT) return;
  int mt = wid / NT, nt = wid % NT;
  v8f acc = wmma_bf16_tile(u_bf, DI, xproj_bf, DI, mt * 16, nt * 16, DI);
  const int lane = threadIdx.x & 31;
  const int col = nt * 16 + (lane & 15);
  const int rbase = (lane < 16) ? 0 : 8;
#pragma unroll
  for (int r = 0; r < 8; ++r)
    dbl[(size_t)(mt * 16 + rbase + r) * DBL_LD + col] = acc[r];
}

// delta = softplus(dbl[:, :12] @ dt_proj_w^T + dt_proj_b)
__global__ void ss1d_delta(const float* __restrict__ dbl, const float* __restrict__ dtw,
                           const float* __restrict__ dtb, float* __restrict__ delta) {
  int idx = blockIdx.x * blockDim.x + threadIdx.x;
  if (idx >= M_ * DI) return;
  int m = idx / DI, d = idx % DI;
  const float* __restrict__ row = dbl + (size_t)m * DBL_LD;
  float s = dtb[d];
#pragma unroll
  for (int r = 0; r < R_; ++r) s = fmaf(row[r], dtw[d * R_ + r], s);
  delta[idx] = (s > 20.0f) ? s : log1pf(__expf(s));
}

// Scan phase 1: per-chunk local scan with h_in = 0. Emits partial y,
// chunk-final local state hL[b,c,d,n], and sum of delta over the chunk.
__global__ void ss1d_scan1(const float* __restrict__ delta, const float* __restrict__ u_f,
                           const float* __restrict__ dbl, const float* __restrict__ Aneg,
                           const float* __restrict__ Ds, float* __restrict__ y,
                           float* __restrict__ hL, float* __restrict__ sdelta) {
  const int bc = blockIdx.x;          // b*NCH + c
  const int d = threadIdx.x;          // 0..383
  float A[N_], h[N_];
#pragma unroll
  for (int n = 0; n < N_; ++n) { A[n] = Aneg[d * N_ + n]; h[n] = 0.0f; }
  const float Dd = Ds[d];
  const size_t base = (size_t)bc * CHL;  // = b*L + c*CHL
  float sdel = 0.0f;
  for (int t = 0; t < CHL; ++t) {
    const size_t m = base + t;
    float dt = delta[m * DI + d];
    float ut = u_f[m * DI + d];
    const float* __restrict__ bcrow = dbl + m * DBL_LD;  // wave-uniform -> scalar loads
    float du = dt * ut;
    float yt = 0.0f;
    sdel += dt;
#pragma unroll
    for (int n = 0; n < N_; ++n) {
      float dA = __expf(dt * A[n]);
      h[n] = fmaf(dA, h[n], du * bcrow[R_ + n]);
      yt = fmaf(h[n], bcrow[R_ + N_ + n], yt);
    }
    y[m * DI + d] = yt + Dd * ut;
    if (t + 4 < CHL) {  // global_prefetch_b8 the streaming operands
      __builtin_prefetch(&delta[(m + 4) * DI + d], 0, 0);
      __builtin_prefetch(&u_f[(m + 4) * DI + d], 0, 0);
    }
  }
#pragma unroll
  for (int n = 0; n < N_; ++n) hL[((size_t)bc * DI + d) * N_ + n] = h[n];
  sdelta[(size_t)bc * DI + d] = sdel;
}

// Scan phase 2: sequential combine across chunks (tiny). Chunk transition
// operator is diagonal: P_n = exp(A_n * sum_delta_chunk).
__global__ void ss1d_scan2(const float* __restrict__ hL, const float* __restrict__ sdelta,
                           const float* __restrict__ Aneg, float* __restrict__ hin) {
  const int b = blockIdx.x;
  const int d = threadIdx.x;
  float A[N_], h[N_];
#pragma unroll
  for (int n = 0; n < N_; ++n) { A[n] = Aneg[d * N_ + n]; h[n] = 0.0f; }
  for (int c = 0; c < NCH; ++c) {
    const size_t o = ((size_t)(b * NCH + c) * DI + d) * N_;
    float sd = sdelta[(size_t)(b * NCH + c) * DI + d];
#pragma unroll
    for (int n = 0; n < N_; ++n) {
      hin[o + n] = h[n];
      h[n] = fmaf(__expf(sd * A[n]), h[n], hL[o + n]);
    }
  }
}

// Scan phase 3: correction  y_t += sum_n C_t[n] * exp(A_n * cumdelta_t) * h_in[n]
__global__ void ss1d_scan3(const float* __restrict__ delta, const float* __restrict__ dbl,
                           const float* __restrict__ Aneg, const float* __restrict__ hin,
                           float* __restrict__ y) {
  const int bc = blockIdx.x;
  if ((bc % NCH) == 0) return;        // first chunk of each batch has h_in = 0
  const int d = threadIdx.x;
  float A[N_], h0[N_];
  const size_t o = ((size_t)bc * DI + d) * N_;
#pragma unroll
  for (int n = 0; n < N_; ++n) { A[n] = Aneg[d * N_ + n]; h0[n] = hin[o + n]; }
  const size_t base = (size_t)bc * CHL;
  float cum = 0.0f;
  for (int t = 0; t < CHL; ++t) {
    const size_t m = base + t;
    cum += delta[m * DI + d];
    const float* __restrict__ bcrow = dbl + m * DBL_LD;
    float add = 0.0f;
#pragma unroll
    for (int n = 0; n < N_; ++n)
      add = fmaf(bcrow[R_ + N_ + n], __expf(cum * A[n]) * h0[n], add);
    y[m * DI + d] += add;
  }
}

// LayerNorm over DI + gate by z; one wave per row (12 elems/lane, wave32 butterfly).
__global__ void ss1d_ln_gate(const float* __restrict__ y, const float* __restrict__ z_f,
                             const float* __restrict__ gamma, const float* __restrict__ beta,
                             bf16* __restrict__ g_bf) {
  int row = (blockIdx.x * blockDim.x + threadIdx.x) >> 5;
  int lane = threadIdx.x & 31;
  if (row >= M_) return;
  const float* __restrict__ yr = y + (size_t)row * DI;
  float vals[12], s1 = 0.0f, s2 = 0.0f;
#pragma unroll
  for (int i = 0; i < 12; ++i) {
    float v = yr[lane + 32 * i];
    vals[i] = v; s1 += v; s2 += v * v;
  }
#pragma unroll
  for (int off = 16; off >= 1; off >>= 1) {
    s1 += __shfl_xor(s1, off, 32);
    s2 += __shfl_xor(s2, off, 32);
  }
  float mu = s1 * (1.0f / DI);
  float var = s2 * (1.0f / DI) - mu * mu;
  float rstd = rsqrtf(var + LN_EPS);
  const float* __restrict__ zr = z_f + (size_t)row * DI;
#pragma unroll
  for (int i = 0; i < 12; ++i) {
    int c = lane + 32 * i;
    float g = fmaf((vals[i] - mu) * rstd, gamma[c], beta[c]) * zr[c];
    g_bf[(size_t)row * DI + c] = (bf16)g;
  }
}

// GEMM3: out = g @ W_out^T  (16384 x 384 @ 384 x 192)
__global__ void ss1d_gemm4(const bf16* __restrict__ g_bf, const bf16* __restrict__ wout_bf,
                           float* __restrict__ out) {
  const int NT = DM / 16;  // 12
  int wid = (blockIdx.x * blockDim.x + threadIdx.x) >> 5;
  if (wid >= (M_ / 16) * NT) return;
  int mt = wid / NT, nt = wid % NT;
  v8f acc = wmma_bf16_tile(g_bf, DI, wout_bf, DI, mt * 16, nt * 16, DI);
  const int lane = threadIdx.x & 31;
  const int col = nt * 16 + (lane & 15);
  const int rbase = (lane < 16) ? 0 : 8;
#pragma unroll
  for (int r = 0; r < 8; ++r)
    out[(size_t)(mt * 16 + rbase + r) * DM + col] = acc[r];
}

// ---------------------------------------------------------------------------
extern "C" void kernel_launch(void* const* d_in, const int* in_sizes, int n_in,
                              void* d_out, int out_size, void* d_ws, size_t ws_size,
                              hipStream_t stream) {
  const float* x        = (const float*)d_in[0];
  const float* W_in     = (const float*)d_in[1];
  const float* x_proj_w = (const float*)d_in[2];
  const float* dt_w     = (const float*)d_in[3];
  const float* dt_b     = (const float*)d_in[4];
  const float* A_logs   = (const float*)d_in[5];
  const float* Ds       = (const float*)d_in[6];
  const float* ln_g     = (const float*)d_in[7];
  const float* ln_b     = (const float*)d_in[8];
  const float* W_out    = (const float*)d_in[9];
  float* out = (float*)d_out;

  // Carve workspace (256B aligned slices)
  char* p = (char*)d_ws;
  auto carve = [&](size_t bytes) -> void* {
    void* r = (void*)p;
    p += (bytes + 255) & ~(size_t)255;
    return r;
  };
  bf16*  x_bf     = (bf16*) carve((size_t)M_ * DM * 2);
  bf16*  win_bf   = (bf16*) carve((size_t)NC1 * DM * 2);
  bf16*  xproj_bf = (bf16*) carve((size_t)48 * DI * 2);
  bf16*  wout_bf  = (bf16*) carve((size_t)DM * DI * 2);
  float* Aneg     = (float*)carve((size_t)DI * N_ * 4);
  float* u_f      = (float*)carve((size_t)M_ * DI * 4);
  bf16*  u_bf     = (bf16*) carve((size_t)M_ * DI * 2);
  float* z_f      = (float*)carve((size_t)M_ * DI * 4);
  float* dbl      = (float*)carve((size_t)M_ * DBL_LD * 4);
  float* delta    = (float*)carve((size_t)M_ * DI * 4);
  float* ybuf     = (float*)carve((size_t)M_ * DI * 4);
  bf16*  g_bf     = (bf16*) carve((size_t)M_ * DI * 2);
  float* hL       = (float*)carve((size_t)B_ * NCH * DI * N_ * 4);
  float* hin      = (float*)carve((size_t)B_ * NCH * DI * N_ * 4);
  float* sdelta   = (float*)carve((size_t)B_ * NCH * DI * 4);

  ss1d_prep<<<(M_ * DM) / 256, 256, 0, stream>>>(x, W_in, x_proj_w, A_logs, W_out,
                                                 x_bf, win_bf, xproj_bf, Aneg, wout_bf);
  ss1d_gemm1<<<((M_ / 16) * (NC1 / 16)) / 8, 256, 0, stream>>>(x_bf, win_bf, u_f, u_bf, z_f);
  ss1d_gemm2<<<((M_ / 16) * (DBL_LD / 16)) / 8, 256, 0, stream>>>(u_bf, xproj_bf, dbl);
  ss1d_delta<<<(M_ * DI) / 256, 256, 0, stream>>>(dbl, dt_w, dt_b, delta);
  ss1d_scan1<<<B_ * NCH, DI, 0, stream>>>(delta, u_f, dbl, Aneg, Ds, ybuf, hL, sdelta);
  ss1d_scan2<<<B_, DI, 0, stream>>>(hL, sdelta, Aneg, hin);
  ss1d_scan3<<<B_ * NCH, DI, 0, stream>>>(delta, dbl, Aneg, hin, ybuf);
  ss1d_ln_gate<<<M_ / 8, 256, 0, stream>>>(ybuf, z_f, ln_g, ln_b, g_bf);
  ss1d_gemm4<<<((M_ / 16) * (DM / 16)) / 8, 256, 0, stream>>>(g_bf, wout_bf, out);
}
